// FixedNeighborSearch_26568667693796
// MI455X (gfx1250) — compile-verified, matrix operand order checked by reading, not run on hardware
//
#include <hip/hip_runtime.h>
#include <hip/hip_bf16.h>

typedef __attribute__((ext_vector_type(2)))  float    v2f;
typedef __attribute__((ext_vector_type(8)))  float    v8f;
typedef __attribute__((ext_vector_type(16))) _Float16 v16h;
typedef __attribute__((ext_vector_type(2)))  int      v2i;

#define WAVES 8
#define QT    16      // queries per wave (WMMA M)
#define NK    33      // keep 33 smallest; slot 32 is the threshold
#define SLOT  34      // padded stride
#define CHUNK 512     // data points staged to LDS per block (6 KB / buffer)
#define INF_F 3.402823466e+38f

#if __has_builtin(__builtin_amdgcn_wmma_f32_16x16x4_f32)
#define HAVE_WMMA_F32X4 1
#endif

#if defined(__AMDGCN__) && __has_builtin(__builtin_amdgcn_global_load_async_to_lds_b64)
#define HAVE_ASYNC_LDS 1
#endif

__device__ __forceinline__ void wait_async_lds()
{
#if defined(HAVE_ASYNC_LDS)
# if __has_builtin(__builtin_amdgcn_s_wait_asynccnt)
    __builtin_amdgcn_s_wait_asynccnt(0);
# else
    asm volatile("s_wait_asynccnt 0x0" ::: "memory");
# endif
#endif
}

// Copy CHUNK data points (CHUNK*3 floats = 6144 B) into stage buffer b.
// 256 threads x 3 sweeps x b64. base must satisfy base+CHUNK <= nd.
__device__ __forceinline__ void stage_chunk(const float* __restrict__ data,
                                            float (*stg)[CHUNK * 3],
                                            int b, int base, int nd)
{
    const int tid = threadIdx.x;
#pragma unroll
    for (int r = 0; r < 3; ++r) {
        const int f = r * 512 + tid * 2;          // float index, 8B aligned
#if defined(HAVE_ASYNC_LDS)
        __builtin_amdgcn_global_load_async_to_lds_b64(
            (v2i*)(data + (size_t)base * 3 + f),
            (v2i*)(&stg[b][f]), 0, 0);
#else
        const float2 v = *(const float2*)(data + (size_t)base * 3 + f);
        *(float2*)&stg[b][f] = v;
#endif
    }
    // pull the chunk after next toward the caches (global_prefetch_b8)
    if (base + 2 * CHUNK <= nd)
        __builtin_prefetch(data + (size_t)(base + CHUNK) * 3 + tid * 2, 0, 3);
}

// One 16x16 tile: dot products via V_WMMA_F32_16X16X4_F32 (D=3 padded to K=4),
// rank key s = d2 - 2*dot (q2 is row-constant), threshold filter + rare insert.
__device__ __forceinline__ void process_tile(
    float qx, float qy, float qz, bool hi,
    float dx, float dy, float dz, float d2,
    int colbase, int nd, int lane, float thrc[8],
    float* __restrict__ sdw, int* __restrict__ siw, float* __restrict__ thrw)
{
    v8f c = {};
#if defined(HAVE_WMMA_F32X4)
    // A 16x4 f32: lanes 0-15 -> K=0,1 ; lanes 16-31 -> K=2,3 (pad=0). B mirrors.
    v2f a, b;
    a.x = hi ? qz : qx;  a.y = hi ? 0.0f : qy;
    b.x = hi ? dz : dx;  b.y = hi ? 0.0f : dy;
    c = __builtin_amdgcn_wmma_f32_16x16x4_f32(
            false, a, false, b, (short)0, c, false, false);
#else
    v16h a = {}; v16h b = {};
    if (!hi) {
        a[0] = (_Float16)qx; a[1] = (_Float16)qy; a[2] = (_Float16)qz;
        b[0] = (_Float16)dx; b[1] = (_Float16)dy; b[2] = (_Float16)dz;
    }
    c = __builtin_amdgcn_wmma_f32_16x16x32_f16(
            false, a, false, b, (short)0, c, false, false);
#endif

    float s[8];
#pragma unroll
    for (int v = 0; v < 8; ++v) s[v] = d2 - 2.0f * c[v];

    // float-max violation reduction -> v_max3 chain + one ballot on fast path
    float viol = thrc[0] - s[0];
#pragma unroll
    for (int v = 1; v < 8; ++v) viol = fmaxf(viol, thrc[v] - s[v]);

    if (__ballot(viol > 0.0f)) {                 // rare slow path
#pragma unroll
        for (int v = 0; v < 8; ++v) {
            unsigned long long m = __ballot(s[v] < thrc[v]);
            while (m) {
                const int l = __ffsll(m) - 1;
                m &= m - 1;
                const float bs   = __shfl(s[v], l);
                const int   brow = v + ((l >= 16) ? 8 : 0);
                const int   bcol = colbase + (l & 15);
                if (lane == 0 && bcol < nd) {    // serialized sorted-insert, 33 slots
                    float* sd = sdw + brow * SLOT;
                    int*   si = siw + brow * SLOT;
                    if (bs < sd[NK - 1]) {
                        int p = NK - 1;
                        while (p > 0 && sd[p - 1] > bs) {
                            sd[p] = sd[p - 1]; si[p] = si[p - 1]; --p;
                        }
                        sd[p] = bs; si[p] = bcol;
                        thrw[brow] = sd[NK - 1];
                    }
                }
            }
        }
#pragma unroll
        for (int v = 0; v < 8; ++v) thrc[v] = thrw[v + (hi ? 8 : 0)];
    }
}

__global__ __launch_bounds__(256) void knn_topk_kernel(
    const float* __restrict__ data, const float* __restrict__ queries,
    long long* __restrict__ out_idx, int nd, int nq)
{
    __shared__ float s_sd[WAVES][QT][SLOT];
    __shared__ int   s_si[WAVES][QT][SLOT];
    __shared__ float s_thr[WAVES][QT];
    __shared__ float s_stage[2][CHUNK * 3];

    const int  lane   = threadIdx.x & 31;
    const int  wave   = threadIdx.x >> 5;
    const int  qbase  = (blockIdx.x * WAVES + wave) * QT;
    const bool active = (qbase < nq);            // inactive waves still stage+barrier

    float* __restrict__ sdw  = &s_sd[wave][0][0];
    int*   __restrict__ siw  = &s_si[wave][0][0];
    float* __restrict__ thrw = &s_thr[wave][0];

    if (active) {
        for (int t = lane; t < QT * SLOT; t += 32) { sdw[t] = INF_F; siw[t] = nd; }
        if (lane < QT) thrw[lane] = INF_F;
    }

    const bool hi   = lane >= 16;
    const int  qrow = lane & 15;
    const int  q    = qbase + qrow;
    const int  qc   = (q < nq) ? q : 0;
    const float qx = queries[3 * qc + 0];
    const float qy = queries[3 * qc + 1];
    const float qz = queries[3 * qc + 2];

    float thrc[8];
#pragma unroll
    for (int v = 0; v < 8; ++v) thrc[v] = INF_F;

    // ---- main pass: double-buffered LDS staging (async-to-LDS when available) ----
    const int nchunks = nd / CHUNK;
    if (nchunks > 0) stage_chunk(data, s_stage, 0, 0, nd);

    for (int cI = 0; cI < nchunks; ++cI) {
        wait_async_lds();                        // s_wait_asynccnt 0
        __syncthreads();                         // chunk cI resident & visible
        if (cI + 1 < nchunks)
            stage_chunk(data, s_stage, (cI + 1) & 1, (cI + 1) * CHUNK, nd);

        if (active) {
            const float* st    = &s_stage[cI & 1][0];
            const int    cbase = cI * CHUNK;
            // register double-buffer the LDS tile reads
            float cx = st[qrow * 3 + 0], cy = st[qrow * 3 + 1], cz = st[qrow * 3 + 2];
            for (int t = 0; t < CHUNK / 16; ++t) {
                const float dx = cx, dy = cy, dz = cz;
                const int nt = (t + 1 < CHUNK / 16) ? (t + 1) * 48 + qrow * 3
                                                    : qrow * 3;
                cx = st[nt + 0]; cy = st[nt + 1]; cz = st[nt + 2];
                const float d2 = dx * dx + dy * dy + dz * dz;
                process_tile(qx, qy, qz, hi, dx, dy, dz, d2,
                             cbase + t * 16, nd, lane, thrc, sdw, siw, thrw);
            }
        }
        __syncthreads();                         // all readers done before reuse
    }

    // ---- tail (nd not a multiple of CHUNK): direct global loads ----
    if (active) {
        for (int col0 = nchunks * CHUNK; col0 < nd; col0 += 16) {
            const int j  = col0 + qrow;
            const int jc = (j < nd) ? j : (nd - 1);
            const float dx = data[3 * jc + 0];
            const float dy = data[3 * jc + 1];
            const float dz = data[3 * jc + 2];
            const float d2 = (j < nd) ? (dx * dx + dy * dy + dz * dz) : INF_F;
            process_tile(qx, qy, qz, hi, dx, dy, dz, d2,
                         col0, nd, lane, thrc, sdw, siw, thrw);
        }
    }

    // ---- emit: per query, bitonic-sort the 32 nearest indices ascending ----
    if (active) {
        for (int r = 0; r < QT; ++r) {
            int idx = s_si[wave][r][lane];       // slots 0..31 (below threshold)
#pragma unroll
            for (int k = 2; k <= 32; k <<= 1) {
#pragma unroll
                for (int jj = k >> 1; jj > 0; jj >>= 1) {
                    const int  o     = __shfl_xor(idx, jj);
                    const bool lower = ((lane & jj) == 0);
                    const bool up    = ((lane & k) == 0);
                    idx = (lower == up) ? (idx < o ? idx : o) : (idx > o ? idx : o);
                }
            }
            out_idx[(long long)(qbase + r) * 32 + lane] = (long long)idx;
        }
    }
}

// splits[i] = 32*i  (counts are generically exactly 32 per query)
__global__ void knn_splits_kernel(long long* __restrict__ out, int nq)
{
    const int i = blockIdx.x * blockDim.x + threadIdx.x;
    if (i <= nq) out[i] = 32LL * (long long)i;
}

extern "C" void kernel_launch(void* const* d_in, const int* in_sizes, int n_in,
                              void* d_out, int out_size, void* d_ws, size_t ws_size,
                              hipStream_t stream)
{
    const float* data    = (const float*)d_in[0];
    const float* queries = (const float*)d_in[1];
    const int nd = in_sizes[0] / 3;
    const int nq = in_sizes[1] / 3;

    long long* out_idx    = (long long*)d_out;            // nq*32 int64
    long long* out_splits = out_idx + (long long)nq * 32; // nq+1  int64

    const int qtiles = (nq + QT - 1) / QT;
    const int blocks = (qtiles + WAVES - 1) / WAVES;
    knn_topk_kernel<<<blocks, 256, 0, stream>>>(data, queries, out_idx, nd, nq);
    knn_splits_kernel<<<(nq + 256) / 256, 256, 0, stream>>>(out_splits, nq);
}
